// DecoderRNN_27032524161364
// MI455X (gfx1250) — compile-verified, matrix operand order checked by reading, not run on hardware
//
#include <hip/hip_runtime.h>
#include <hip/hip_bf16.h>

// ---------------------------------------------------------------------------
// Problem constants (from reference): T=32, B=256, DIN=4196, H=1024, C=151
// ---------------------------------------------------------------------------
#define T_STEPS 32
#define B_SZ    256
#define DIN_SZ  4196
#define H_SZ    1024
#define C_SZ    151

#define KP_IN   4224   // DIN padded to multiple of 32 for bf16 WMMA K-loop
#define C_PAD   192    // C padded to multiple of 64 for output-GEMM tiling
#define TB      (T_STEPS * B_SZ)   // 8192

typedef __bf16 bf16;
typedef __attribute__((ext_vector_type(16))) __bf16 v16bf;
typedef __attribute__((ext_vector_type(8)))  float  v8f;

union FragU { v16bf v; uint4 q[2]; };

// Load one 16x32 bf16 WMMA fragment (A or B operand) given the row base
// pointer. Per CDNA5 ISA layout for 16-bit A/B fragments:
//   lanes 0-15 : row = lane,    halves 0-7 = K[k0..k0+7],  halves 8-15 = K[k0+16..+23]
//   lanes 16-31: row = lane-16, halves 0-7 = K[k0+8..+15], halves 8-15 = K[k0+24..+31]
__device__ __forceinline__ v16bf load_frag_row(const bf16* __restrict__ rowp, int k) {
    FragU f;
    f.q[0] = *(const uint4*)(rowp + k);
    f.q[1] = *(const uint4*)(rowp + k + 16);
    return f.v;
}

// ---------------------------------------------------------------------------
// Generic bf16 WMMA GEMM:  D[M,N] = A[M,Kp] * B[N,Kp]^T + bias[N]
//   block = 128 threads (4 waves, arranged 2x2)
//   wave tile = (MT*16) x (NT*16); block tile = (MT*32) x (NT*32)
//   grid = (M / (MT*32), N / (NT*32))
// ---------------------------------------------------------------------------
template <int MT, int NT>
__global__ void __launch_bounds__(128)
gemm_bf16_wmma(const bf16* __restrict__ A, const bf16* __restrict__ Bm,
               const float* __restrict__ bias, float* __restrict__ D,
               int Kp, int ldd, int Nvalid) {
    const int lane = threadIdx.x & 31;
    const int wave = threadIdx.x >> 5;          // 0..3
    const int m0 = blockIdx.x * (MT * 32) + (wave >> 1) * (MT * 16);
    const int n0 = blockIdx.y * (NT * 32) + (wave & 1) * (NT * 16);

    const int frow = lane & 15;                 // row within a 16-wide tile
    const int hi   = lane >> 4;
    const int koff = hi << 3;                   // fragment K sub-offset per lane half

    // init accumulators with bias (bias depends only on column n)
    v8f acc[MT][NT];
#pragma unroll
    for (int ni = 0; ni < NT; ++ni) {
        const int n = n0 + ni * 16 + frow;
        const float bv = (bias && n < Nvalid) ? bias[n] : 0.0f;
#pragma unroll
        for (int mi = 0; mi < MT; ++mi)
#pragma unroll
            for (int v = 0; v < 8; ++v) acc[mi][ni][v] = bv;
    }

    // hoist per-row base pointers out of the K loop
    const bf16* aRow[MT];
    const bf16* bRow[NT];
#pragma unroll
    for (int mi = 0; mi < MT; ++mi)
        aRow[mi] = A + (size_t)(m0 + mi * 16 + frow) * (size_t)Kp + koff;
#pragma unroll
    for (int ni = 0; ni < NT; ++ni)
        bRow[ni] = Bm + (size_t)(n0 + ni * 16 + frow) * (size_t)Kp + koff;

    for (int k0 = 0; k0 < Kp; k0 += 32) {
        if (k0 + 32 < Kp) {                     // emits global_prefetch_b8
            __builtin_prefetch(aRow[0] + k0 + 32, 0, 1);
            __builtin_prefetch(bRow[0] + k0 + 32, 0, 1);
        }
        v16bf af[MT], bfr[NT];
#pragma unroll
        for (int mi = 0; mi < MT; ++mi) af[mi]  = load_frag_row(aRow[mi], k0);
#pragma unroll
        for (int ni = 0; ni < NT; ++ni) bfr[ni] = load_frag_row(bRow[ni], k0);

#pragma unroll
        for (int mi = 0; mi < MT; ++mi)
#pragma unroll
            for (int ni = 0; ni < NT; ++ni)
                acc[mi][ni] = __builtin_amdgcn_wmma_f32_16x16x32_bf16(
                    false, af[mi], false, bfr[ni], (short)0, acc[mi][ni],
                    false, false);
    }

    // store: VGPR v of a 16x16 f32 tile holds (M = v + 8*hi, N = frow)
#pragma unroll
    for (int mi = 0; mi < MT; ++mi)
#pragma unroll
        for (int ni = 0; ni < NT; ++ni) {
            const int n = n0 + ni * 16 + frow;
            if (n < Nvalid) {
#pragma unroll
                for (int v = 0; v < 8; ++v) {
                    const int m = m0 + mi * 16 + v + hi * 8;
                    D[(size_t)m * (size_t)ldd + n] = acc[mi][ni][v];
                }
            }
        }
}

// ---------------------------------------------------------------------------
// f32 -> bf16 staging with zero padding in rows (Rvalid) and K (Kin -> Kp)
// ---------------------------------------------------------------------------
__global__ void cvt_pad_bf16(const float* __restrict__ src, bf16* __restrict__ dst,
                             int R, int Rvalid, int Kin, int Kp) {
    const size_t idx = (size_t)blockIdx.x * blockDim.x + threadIdx.x;
    const size_t total = (size_t)R * Kp;
    if (idx >= total) return;
    const int r = (int)(idx / Kp);
    const int k = (int)(idx % Kp);
    float v = 0.0f;
    if (r < Rvalid && k < Kin) v = src[(size_t)r * Kin + k];
    dst[idx] = (bf16)v;
}

__global__ void init_state(const float* __restrict__ h0, const float* __restrict__ c0,
                           bf16* __restrict__ h_bf, float* __restrict__ c_ws, int n) {
    const int i = blockIdx.x * blockDim.x + threadIdx.x;
    if (i >= n) return;
    h_bf[i] = (bf16)h0[i];
    c_ws[i] = c0[i];
}

// ---------------------------------------------------------------------------
// Gate nonlinearities for one timestep.
//   PI_t : [B, 6H] f32 (input projection slice for step t, b_in already added)
//   PS   : [B, 5H] f32 (recurrent projection, b_state already added)
// Writes c (f32, carried), h_bf (bf16, next step's GEMM A), hs_t (bf16 history)
// ---------------------------------------------------------------------------
__device__ __forceinline__ float sigmoidf_(float x) {
    return 1.0f / (1.0f + __expf(-x));
}

__global__ void gate_kernel(const float* __restrict__ PI_t, const float* __restrict__ PS,
                            const float* __restrict__ drop, float* __restrict__ c,
                            bf16* __restrict__ h_bf, bf16* __restrict__ hs_t) {
    const int idx = blockIdx.x * blockDim.x + threadIdx.x;
    if (idx >= B_SZ * H_SZ) return;
    const int b = idx >> 10;          // / H_SZ
    const int j = idx & (H_SZ - 1);   // % H_SZ

    const float* pi = PI_t + (size_t)b * (6 * H_SZ);
    const float* ps = PS   + (size_t)b * (5 * H_SZ);

    const float i_g = sigmoidf_(pi[0 * H_SZ + j] + ps[0 * H_SZ + j]);
    const float f_g = sigmoidf_(pi[1 * H_SZ + j] + ps[1 * H_SZ + j]);
    const float m_i = tanhf    (pi[2 * H_SZ + j] + ps[2 * H_SZ + j]);
    const float o_g = sigmoidf_(pi[3 * H_SZ + j] + ps[3 * H_SZ + j]);
    const float mem = i_g * m_i + f_g * c[idx];
    float out = o_g * tanhf(mem);
    const float hw = sigmoidf_(pi[4 * H_SZ + j] + ps[4 * H_SZ + j]);
    out = hw * out + (1.0f - hw) * pi[5 * H_SZ + j];
    out *= drop[idx];

    c[idx]    = mem;
    h_bf[idx] = (bf16)out;
    hs_t[idx] = (bf16)out;
}

// ---------------------------------------------------------------------------
// Host-side orchestration
// ---------------------------------------------------------------------------
extern "C" void kernel_launch(void* const* d_in, const int* in_sizes, int n_in,
                              void* d_out, int out_size, void* d_ws, size_t ws_size,
                              hipStream_t stream) {
    const float* x       = (const float*)d_in[0];   // [T,B,DIN]
    const float* h0      = (const float*)d_in[1];   // [B,H]
    const float* c0      = (const float*)d_in[2];   // [B,H]
    const float* drop    = (const float*)d_in[3];   // [B,H]
    const float* W_in    = (const float*)d_in[4];   // [6H,DIN]
    const float* b_in    = (const float*)d_in[5];   // [6H]
    const float* W_state = (const float*)d_in[6];   // [5H,H]
    const float* b_state = (const float*)d_in[7];   // [5H]
    const float* W_out   = (const float*)d_in[8];   // [C,H]
    const float* b_out   = (const float*)d_in[9];   // [C]

    char* ws = (char*)d_ws;
    size_t off = 0;
    auto take = [&](size_t bytes) { char* p = ws + off; off += (bytes + 255) & ~(size_t)255; return p; };

    bf16*  Xbf    = (bf16*) take((size_t)TB * KP_IN * 2);          //  69.2 MB
    bf16*  Winbf  = (bf16*) take((size_t)6 * H_SZ * KP_IN * 2);    //  51.9 MB
    bf16*  Wstbf  = (bf16*) take((size_t)5 * H_SZ * H_SZ * 2);     //  10.5 MB
    bf16*  Woutbf = (bf16*) take((size_t)C_PAD * H_SZ * 2);        //   0.4 MB
    float* PI     = (float*)take((size_t)TB * 6 * H_SZ * 4);       // 201.3 MB
    float* PS     = (float*)take((size_t)B_SZ * 5 * H_SZ * 4);     //   5.2 MB
    bf16*  h_bf   = (bf16*) take((size_t)B_SZ * H_SZ * 2);         //   0.5 MB
    float* c_ws   = (float*)take((size_t)B_SZ * H_SZ * 4);         //   1.0 MB
    bf16*  hs_bf  = (bf16*) take((size_t)TB * H_SZ * 2);           //  16.8 MB

    // ---- stage inputs/weights to bf16 (zero-padded) ----
    {
        size_t n = (size_t)TB * KP_IN;
        cvt_pad_bf16<<<(unsigned)((n + 255) / 256), 256, 0, stream>>>(x, Xbf, TB, TB, DIN_SZ, KP_IN);
        n = (size_t)6 * H_SZ * KP_IN;
        cvt_pad_bf16<<<(unsigned)((n + 255) / 256), 256, 0, stream>>>(W_in, Winbf, 6 * H_SZ, 6 * H_SZ, DIN_SZ, KP_IN);
        n = (size_t)5 * H_SZ * H_SZ;
        cvt_pad_bf16<<<(unsigned)((n + 255) / 256), 256, 0, stream>>>(W_state, Wstbf, 5 * H_SZ, 5 * H_SZ, H_SZ, H_SZ);
        n = (size_t)C_PAD * H_SZ;
        cvt_pad_bf16<<<(unsigned)((n + 255) / 256), 256, 0, stream>>>(W_out, Woutbf, C_PAD, C_SZ, H_SZ, H_SZ);
        init_state<<<(B_SZ * H_SZ + 255) / 256, 256, 0, stream>>>(h0, c0, h_bf, c_ws, B_SZ * H_SZ);
    }

    // ---- hoisted input projection: PI[TB, 6H] = X @ W_in^T + b_in ----
    // 128x128 block tiles: high arithmetic intensity (1 b128 load per WMMA)
    gemm_bf16_wmma<4, 4><<<dim3(TB / 128, (6 * H_SZ) / 128), 128, 0, stream>>>(
        Xbf, Winbf, b_in, PI, KP_IN, 6 * H_SZ, 6 * H_SZ);

    // ---- sequential recurrence ----
    for (int t = 0; t < T_STEPS; ++t) {
        // PS[B,5H] = h @ W_state^T + b_state  (64x64 blocks -> 320 WGs of parallelism)
        gemm_bf16_wmma<2, 2><<<dim3(B_SZ / 64, (5 * H_SZ) / 64), 128, 0, stream>>>(
            h_bf, Wstbf, b_state, PS, H_SZ, 5 * H_SZ, 5 * H_SZ);
        // gates -> h(t), c(t), hs[t]
        gate_kernel<<<(B_SZ * H_SZ + 255) / 256, 256, 0, stream>>>(
            PI + (size_t)t * B_SZ * 6 * H_SZ, PS, drop, c_ws,
            h_bf, hs_bf + (size_t)t * B_SZ * H_SZ);
    }

    // ---- output projection: logits[TB, C] = hs @ W_out^T + b_out ----
    gemm_bf16_wmma<4, 2><<<dim3(TB / 128, C_PAD / 64), 128, 0, stream>>>(
        hs_bf, Woutbf, b_out, (float*)d_out, H_SZ, C_SZ, C_SZ);
}
